// StableAttention_29652454211601
// MI455X (gfx1250) — compile-verified, matrix operand order checked by reading, not run on hardware
//
#include <hip/hip_runtime.h>

#define D_MODEL 1024
#define N_HEADS 16
#define D_K 64
#define SEQ 2048
#define BATCH 2
#define ROWS (BATCH*SEQ)          // 4096
#define SCALE_C 0.125f            // 1/sqrt(64)
#define MASK_VAL -1.0e9f
#define EPS_FP32 1e-8f

typedef __attribute__((ext_vector_type(16))) __bf16 v16bf;
typedef __attribute__((ext_vector_type(8)))  float  v8f;

union BfOp { v16bf v; unsigned u[8]; };

__device__ inline unsigned short f2bf(float f) {
    unsigned x = __float_as_uint(f);
    unsigned r = x + 0x7FFFu + ((x >> 16) & 1u);   // round-to-nearest-even
    return (unsigned short)(r >> 16);
}
__device__ inline unsigned pack2(float a, float b) {
#if __has_builtin(__builtin_amdgcn_cvt_pk_bf16_f32)
    auto r = __builtin_amdgcn_cvt_pk_bf16_f32(a, b);   // v_cvt_pk_bf16_f32
    unsigned u; __builtin_memcpy(&u, &r, sizeof(u));
    return u;
#else
    return (unsigned)f2bf(a) | ((unsigned)f2bf(b) << 16);
#endif
}

// WMMA bf16 operand loader from a row of bf16 elements (uint-aligned rows).
// A-layout (16x32, M per lane): lanes 0-15 hold K = {k0+0..7, k0+16..23},
// lanes 16-31 hold K = {k0+8..15, k0+24..31}.  B-layout is symmetric (N per lane).
__device__ inline v16bf load_bf16_row(const unsigned short* row, int k0, int half) {
    const unsigned* p = (const unsigned*)row;
    BfOp o;
    int o1 = (k0 + half * 8) >> 1;
    int o2 = (k0 + 16 + half * 8) >> 1;
#pragma unroll
    for (int t = 0; t < 4; ++t) o.u[t]     = p[o1 + t];
#pragma unroll
    for (int t = 0; t < 4; ++t) o.u[4 + t] = p[o2 + t];
    return o.v;
}

__device__ inline v8f wmma_bf16(v16bf a, v16bf b, v8f c) {
    return __builtin_amdgcn_wmma_f32_16x16x32_bf16(false, a, false, b, (short)0, c, false, false);
}

// ---------------- Kernel 0: one-time fp32 -> bf16 conversion ---------------
__global__ __launch_bounds__(256) void cvt_bf16_kernel(
    const float* __restrict__ src, unsigned short* __restrict__ dst, int n4)
{
    int i = blockIdx.x * blockDim.x + threadIdx.x;
    int stride = gridDim.x * blockDim.x;
    const float4* s4 = (const float4*)src;
    uint2* d2 = (uint2*)dst;
    for (; i < n4; i += stride) {
        float4 v = s4[i];
        d2[i] = make_uint2(pack2(v.x, v.y), pack2(v.z, v.w));
    }
}

// ---------------- Kernel 1: fused Q/K/V projection  y = x @ W^T + b --------
// grid (512, 3), block 256 (8 waves); each wave -> one 16x64 output tile.
// k-loop unrolled x2 with alternating operand buffers (no rotation copies).
__global__ __launch_bounds__(256) void qkv_proj_kernel(
    const unsigned short* __restrict__ xb,
    const unsigned short* __restrict__ Wqb, const float* __restrict__ bq,
    const unsigned short* __restrict__ Wkb, const float* __restrict__ bk,
    const unsigned short* __restrict__ Wvb, const float* __restrict__ bv,
    unsigned short* __restrict__ Qb, unsigned short* __restrict__ Kbuf,
    unsigned short* __restrict__ Vbuf)
{
    const int lane = threadIdx.x & 31;
    const int wave = threadIdx.x >> 5;
    const int half = lane >> 4;
    const int l15  = lane & 15;

    const unsigned short* W; const float* bias; unsigned short* dst;
    if (blockIdx.y == 0)      { W = Wqb; bias = bq; dst = Qb;   }
    else if (blockIdx.y == 1) { W = Wkb; bias = bk; dst = Kbuf; }
    else                      { W = Wvb; bias = bv; dst = Vbuf; }

    const int tile = blockIdx.x * 8 + wave;    // 4096 wave-tiles
    const int row0 = (tile >> 4) * 16;         // 16 col-blocks of width 64
    const int cb   = tile & 15;                // head index (64-wide)
    const int col0 = cb * 64;

    const unsigned short* arow  = xb + (size_t)(row0 + l15) * D_MODEL;
    const unsigned short* brow0 = W  + (size_t)(col0 + l15) * D_MODEL;

    v8f acc[4];
#pragma unroll
    for (int t = 0; t < 4; ++t) acc[t] = (v8f){};

    v16bf a0, a1, b0[4], b1[4];
    a0 = load_bf16_row(arow, 0, half);
#pragma unroll
    for (int t = 0; t < 4; ++t)
        b0[t] = load_bf16_row(brow0 + (size_t)t * 16 * D_MODEL, 0, half);

    int k0 = 0;
    for (; k0 + 64 < D_MODEL; k0 += 64) {
        a1 = load_bf16_row(arow, k0 + 32, half);
#pragma unroll
        for (int t = 0; t < 4; ++t)
            b1[t] = load_bf16_row(brow0 + (size_t)t * 16 * D_MODEL, k0 + 32, half);
#pragma unroll
        for (int t = 0; t < 4; ++t) acc[t] = wmma_bf16(a0, b0[t], acc[t]);

        a0 = load_bf16_row(arow, k0 + 64, half);
#pragma unroll
        for (int t = 0; t < 4; ++t)
            b0[t] = load_bf16_row(brow0 + (size_t)t * 16 * D_MODEL, k0 + 64, half);
#pragma unroll
        for (int t = 0; t < 4; ++t) acc[t] = wmma_bf16(a1, b1[t], acc[t]);
    }
    // tail: k0 == D_MODEL-64; consume buf0 (k0) and buf1 (k0+32)
    a1 = load_bf16_row(arow, k0 + 32, half);
#pragma unroll
    for (int t = 0; t < 4; ++t)
        b1[t] = load_bf16_row(brow0 + (size_t)t * 16 * D_MODEL, k0 + 32, half);
#pragma unroll
    for (int t = 0; t < 4; ++t) acc[t] = wmma_bf16(a0, b0[t], acc[t]);
#pragma unroll
    for (int t = 0; t < 4; ++t) acc[t] = wmma_bf16(a1, b1[t], acc[t]);

#pragma unroll
    for (int t = 0; t < 4; ++t) {
        const int d  = t * 16 + l15;           // within-head dim
        const float bb = bias[col0 + d];
#pragma unroll
        for (int r = 0; r < 8; ++r) {
            int grow = row0 + r + half * 8;    // global row in [0,4096)
            int bidx = grow >> 11;             // / SEQ
            int s    = grow & (SEQ - 1);
            size_t addr = (((size_t)(bidx * N_HEADS + cb)) * SEQ + s) * D_K + d;
            dst[addr] = f2bf(acc[t][r] + bb);
        }
    }
}

// ---------------- Kernel 2: causal flash attention -------------------------
// grid (S/128, B*H), block 256.  Wave w owns query rows [q0blk + 16w, +16).
__global__ __launch_bounds__(256) void attn_kernel(
    const unsigned short* __restrict__ Qb,
    const unsigned short* __restrict__ Kb,
    const unsigned short* __restrict__ Vb,
    unsigned short* __restrict__ Ob)
{
    __shared__ unsigned short Kl[32 * 64];        // key block, row-major
    __shared__ unsigned short Vt[64 * 34];        // value block, transposed (+pad)
    __shared__ unsigned short Pl[8][16 * 32];     // per-wave P staging

    const int tid  = threadIdx.x;
    const int lane = tid & 31;
    const int wave = tid >> 5;
    const int half = lane >> 4;
    const int l15  = lane & 15;

    const int bh    = blockIdx.y;                 // b*H + h
    const int q0blk = blockIdx.x * 128;
    const int q0    = q0blk + wave * 16;

    const unsigned short* Qg = Qb + (size_t)bh * SEQ * D_K;
    const unsigned short* Kg = Kb + (size_t)bh * SEQ * D_K;
    const unsigned short* Vg = Vb + (size_t)bh * SEQ * D_K;

    // Q A-operands for this wave's 16 rows (Dk=64 -> two K=32 steps)
    const unsigned short* qrow = Qg + (size_t)(q0 + l15) * D_K;
    v16bf aQ0 = load_bf16_row(qrow, 0, half);
    v16bf aQ1 = load_bf16_row(qrow, 32, half);

    // LDS byte offset of the K staging buffer (low 32 bits of generic ptr)
    const unsigned kl_lds = (unsigned)(size_t)(void*)Kl + (unsigned)tid * 16u;

    float m_run[8], l_run[8];
    v8f accO[4];
#pragma unroll
    for (int r = 0; r < 8; ++r) { m_run[r] = -3.0e38f; l_run[r] = 0.f; }
#pragma unroll
    for (int t = 0; t < 4; ++t) accO[t] = (v8f){};

    const int nkb = (q0blk + 128) >> 5;           // causal: keys < q0blk+128
    for (int kb = 0; kb < nkb; ++kb) {
        const int key0 = kb * 32;
        // ---- K block: async global -> LDS (16B per thread = 4KB total) ----
        {
            const unsigned short* gsrc = Kg + (size_t)key0 * D_K + (size_t)tid * 8;
            asm volatile("global_load_async_to_lds_b128 %0, %1, off"
                         :: "v"(kl_lds), "v"(gsrc) : "memory");
        }
        // ---- V block: staged transposed through VGPRs ----
        {
#pragma unroll
            for (int i = 0; i < 8; ++i) {
                int e = i * 256 + tid;            // 0..2047
                int key = e >> 6, d = e & 63;
                Vt[d * 34 + key] = Vg[(size_t)(key0 + key) * D_K + d];
            }
        }
        // prefetch next key block while this one is consumed
        if (kb + 1 < nkb) {
            const unsigned short* nk = Kg + (size_t)(key0 + 32) * D_K;
            const unsigned short* nv = Vg + (size_t)(key0 + 32) * D_K;
            __builtin_prefetch(nk + tid * 8, 0, 1);   // global_prefetch_b8
            __builtin_prefetch(nv + tid * 8, 0, 1);
        }
        asm volatile("s_wait_asynccnt 0" ::: "memory");
        __syncthreads();

        // ---- scores S = Q * K^T for two 16-key subtiles ----
        v8f sacc[2];
#pragma unroll
        for (int nt = 0; nt < 2; ++nt) {
            const unsigned short* krow = Kl + (nt * 16 + l15) * 64;
            v16bf b0 = load_bf16_row(krow, 0, half);
            v16bf b1 = load_bf16_row(krow, 32, half);
            v8f s = {};
            s = wmma_bf16(aQ0, b0, s);
            s = wmma_bf16(aQ1, b1, s);
            sacc[nt] = s;
        }

        // ---- scale + causal mask + row max ----
        float pvals[2][8], rowmax[8];
#pragma unroll
        for (int r = 0; r < 8; ++r) rowmax[r] = -3.0e38f;
#pragma unroll
        for (int nt = 0; nt < 2; ++nt) {
            int key = key0 + nt * 16 + l15;
#pragma unroll
            for (int r = 0; r < 8; ++r) {
                int q = q0 + r + half * 8;
                float sv = sacc[nt][r] * SCALE_C;
                sv = (key > q) ? MASK_VAL : sv;
                pvals[nt][r] = sv;
                rowmax[r] = fmaxf(rowmax[r], sv);
            }
        }
#pragma unroll
        for (int r = 0; r < 8; ++r) {           // reduce over the 16 lanes of a half
            float v = rowmax[r];
            v = fmaxf(v, __shfl_xor(v, 1, 32));
            v = fmaxf(v, __shfl_xor(v, 2, 32));
            v = fmaxf(v, __shfl_xor(v, 4, 32));
            v = fmaxf(v, __shfl_xor(v, 8, 32));
            rowmax[r] = v;
        }

        float alpha[8], rowsum[8];
#pragma unroll
        for (int r = 0; r < 8; ++r) {
            float mnew = fmaxf(m_run[r], rowmax[r]);
            alpha[r] = __expf(m_run[r] - mnew);
            m_run[r] = mnew;
            rowsum[r] = 0.f;
        }

        // ---- P = exp(S - m), staged to LDS (D-layout -> A-layout) ----
        unsigned short* pw = Pl[wave];
#pragma unroll
        for (int nt = 0; nt < 2; ++nt) {
#pragma unroll
            for (int r = 0; r < 8; ++r) {
                float p = __expf(pvals[nt][r] - m_run[r]);
                rowsum[r] += p;
                pw[(r + half * 8) * 32 + nt * 16 + l15] = f2bf(p);
            }
        }
#pragma unroll
        for (int r = 0; r < 8; ++r) {
            float v = rowsum[r];
            v += __shfl_xor(v, 1, 32);
            v += __shfl_xor(v, 2, 32);
            v += __shfl_xor(v, 4, 32);
            v += __shfl_xor(v, 8, 32);
            l_run[r] = l_run[r] * alpha[r] + v;
        }
#pragma unroll
        for (int t = 0; t < 4; ++t)
#pragma unroll
            for (int r = 0; r < 8; ++r) accO[t][r] *= alpha[r];

        // same-wave cross-lane LDS RAW: drain DS before reading P back
        asm volatile("s_wait_dscnt 0" ::: "memory");

        v16bf aP = load_bf16_row(pw + l15 * 32, 0, half);
#pragma unroll
        for (int t = 0; t < 4; ++t) {
            v16bf bvv = load_bf16_row(Vt + (t * 16 + l15) * 34, 0, half);
            accO[t] = wmma_bf16(aP, bvv, accO[t]);
        }
        __syncthreads();
    }

    // ---- epilogue: O[b, q, h*64 + d] = acc / (l + eps)  (bf16) ----
    int b = bh >> 4, h = bh & 15;
#pragma unroll
    for (int r = 0; r < 8; ++r) {
        int q = q0 + r + half * 8;
        float inv = 1.0f / (l_run[r] + EPS_FP32);
        size_t rowbase = ((size_t)b * SEQ + q) * D_MODEL + h * D_K;
#pragma unroll
        for (int t = 0; t < 4; ++t)
            Ob[rowbase + t * 16 + l15] = f2bf(accO[t][r] * inv);
    }
}

// ---------------- Kernel 3: output projection  out = O @ Wo^T + bo ---------
__global__ __launch_bounds__(256) void out_proj_kernel(
    const unsigned short* __restrict__ Ob,
    const unsigned short* __restrict__ Wob, const float* __restrict__ bo,
    float* __restrict__ out)
{
    const int lane = threadIdx.x & 31;
    const int wave = threadIdx.x >> 5;
    const int half = lane >> 4;
    const int l15  = lane & 15;

    const int tile = blockIdx.x * 8 + wave;
    const int row0 = (tile >> 4) * 16;
    const int col0 = (tile & 15) * 64;

    const unsigned short* arow  = Ob  + (size_t)(row0 + l15) * D_MODEL;
    const unsigned short* brow0 = Wob + (size_t)(col0 + l15) * D_MODEL;

    v8f acc[4];
#pragma unroll
    for (int t = 0; t < 4; ++t) acc[t] = (v8f){};

    v16bf a0, a1, b0[4], b1[4];
    a0 = load_bf16_row(arow, 0, half);
#pragma unroll
    for (int t = 0; t < 4; ++t)
        b0[t] = load_bf16_row(brow0 + (size_t)t * 16 * D_MODEL, 0, half);

    int k0 = 0;
    for (; k0 + 64 < D_MODEL; k0 += 64) {
        a1 = load_bf16_row(arow, k0 + 32, half);
#pragma unroll
        for (int t = 0; t < 4; ++t)
            b1[t] = load_bf16_row(brow0 + (size_t)t * 16 * D_MODEL, k0 + 32, half);
#pragma unroll
        for (int t = 0; t < 4; ++t) acc[t] = wmma_bf16(a0, b0[t], acc[t]);

        a0 = load_bf16_row(arow, k0 + 64, half);
#pragma unroll
        for (int t = 0; t < 4; ++t)
            b0[t] = load_bf16_row(brow0 + (size_t)t * 16 * D_MODEL, k0 + 64, half);
#pragma unroll
        for (int t = 0; t < 4; ++t) acc[t] = wmma_bf16(a1, b1[t], acc[t]);
    }
    a1 = load_bf16_row(arow, k0 + 32, half);
#pragma unroll
    for (int t = 0; t < 4; ++t)
        b1[t] = load_bf16_row(brow0 + (size_t)t * 16 * D_MODEL, k0 + 32, half);
#pragma unroll
    for (int t = 0; t < 4; ++t) acc[t] = wmma_bf16(a0, b0[t], acc[t]);
#pragma unroll
    for (int t = 0; t < 4; ++t) acc[t] = wmma_bf16(a1, b1[t], acc[t]);

#pragma unroll
    for (int t = 0; t < 4; ++t) {
        float bb = bo[col0 + t * 16 + l15];
#pragma unroll
        for (int r = 0; r < 8; ++r) {
            int grow = row0 + r + half * 8;
            out[(size_t)grow * D_MODEL + col0 + t * 16 + l15] = acc[t][r] + bb;
        }
    }
}

extern "C" void kernel_launch(void* const* d_in, const int* in_sizes, int n_in,
                              void* d_out, int out_size, void* d_ws, size_t ws_size,
                              hipStream_t stream) {
    (void)in_sizes; (void)n_in; (void)out_size; (void)ws_size;
    const float* x  = (const float*)d_in[0];
    const float* Wq = (const float*)d_in[1];
    const float* bq = (const float*)d_in[2];
    const float* Wk = (const float*)d_in[3];
    const float* bk = (const float*)d_in[4];
    const float* Wv = (const float*)d_in[5];
    const float* bv = (const float*)d_in[6];
    const float* Wo = (const float*)d_in[7];
    const float* bo = (const float*)d_in[8];
    // d_in[9] = mask (causality computed analytically)

    const size_t nelem = (size_t)BATCH * N_HEADS * SEQ * D_K;  // 4M
    const size_t welem = (size_t)D_MODEL * D_MODEL;            // 1M
    unsigned short* Qb  = (unsigned short*)d_ws;
    unsigned short* Kb  = Qb + nelem;
    unsigned short* Vb  = Qb + 2 * nelem;
    unsigned short* Obf = Qb + 3 * nelem;
    unsigned short* xb  = Qb + 4 * nelem;          // 4M elems
    unsigned short* Wqb = xb + nelem;
    unsigned short* Wkb = Wqb + welem;
    unsigned short* Wvb = Wkb + welem;
    unsigned short* Wob = Wvb + welem;

    // one-time conversions (x: 4M, weights: 1M each)
    cvt_bf16_kernel<<<1024, 256, 0, stream>>>(x,  xb,  (int)(nelem / 4));
    cvt_bf16_kernel<<<512, 256, 0, stream>>>(Wq, Wqb, (int)(welem / 4));
    cvt_bf16_kernel<<<512, 256, 0, stream>>>(Wk, Wkb, (int)(welem / 4));
    cvt_bf16_kernel<<<512, 256, 0, stream>>>(Wv, Wvb, (int)(welem / 4));
    cvt_bf16_kernel<<<512, 256, 0, stream>>>(Wo, Wob, (int)(welem / 4));

    qkv_proj_kernel<<<dim3(512, 3), 256, 0, stream>>>(
        xb, Wqb, bq, Wkb, bk, Wvb, bv, Qb, Kb, Vb);

    attn_kernel<<<dim3(SEQ / 128, BATCH * N_HEADS), 256, 0, stream>>>(Qb, Kb, Vb, Obf);

    out_proj_kernel<<<512, 256, 0, stream>>>(Obf, Wob, bo, (float*)d_out);
}
